// TransformerLayer_35158602285745
// MI455X (gfx1250) — matrix operand order repaired and checked by reading, not compiled, not run on hardware
//
#include <hip/hip_runtime.h>

// ---------------------------------------------------------------------------
// CDNA5 WMMA types
// ---------------------------------------------------------------------------
typedef __attribute__((ext_vector_type(16))) __bf16 v16bf;
typedef __attribute__((ext_vector_type(8)))  float  v8f;
typedef __attribute__((ext_vector_type(4)))  int    v4i_t;

union FragBF {
    unsigned u[8];
    v16bf    v;
};
union FragF {
    float f[8];
    v8f   v;
};

__device__ __forceinline__ unsigned short f2bf(float f) {
    unsigned x = __builtin_bit_cast(unsigned, f);
    x += 0x7fffu + ((x >> 16) & 1u);   // round-to-nearest-even
    return (unsigned short)(x >> 16);
}

// ---- async global->LDS staging (gfx1250), with safe fallback --------------
#if defined(__has_builtin)
#if __has_builtin(__builtin_amdgcn_global_load_async_to_lds_b128) && \
    __has_builtin(__builtin_amdgcn_s_wait_asynccnt)
#define HAVE_ASYNC_LDS 1
#endif
#endif
#ifndef HAVE_ASYNC_LDS
#define HAVE_ASYNC_LDS 0
#endif

#if HAVE_ASYNC_LDS
typedef __attribute__((address_space(1))) v4i_t* gptr_b128;  // global int4*
typedef __attribute__((address_space(3))) v4i_t* lptr_b128;  // LDS    int4*
#endif

__device__ __forceinline__ void copy16_g2lds(const unsigned short* src,
                                             unsigned short* lds_dst) {
#if HAVE_ASYNC_LDS
    __builtin_amdgcn_global_load_async_to_lds_b128(
        (gptr_b128)(void*)src, (lptr_b128)(void*)lds_dst, 0, 0);
#else
    *(uint4*)lds_dst = *(const uint4*)src;
#endif
}

__device__ __forceinline__ void wait_async_le3() {
#if HAVE_ASYNC_LDS
    __builtin_amdgcn_s_wait_asynccnt(3);
#endif
}
__device__ __forceinline__ void wait_async_le0() {
#if HAVE_ASYNC_LDS
    __builtin_amdgcn_s_wait_asynccnt(0);
#endif
}

// ---------------------------------------------------------------------------
// f32 -> bf16 conversion (grid-stride)
// ---------------------------------------------------------------------------
__global__ void cvt_f32_bf16(const float* __restrict__ in,
                             unsigned short* __restrict__ out, int n) {
    for (int i = blockIdx.x * blockDim.x + threadIdx.x; i < n;
         i += gridDim.x * blockDim.x) {
        out[i] = f2bf(in[i]);
    }
}

// f32 [K,N] row-major  ->  bf16 transposed [N,K]
__global__ void cvt_f32_bf16_T(const float* __restrict__ in,
                               unsigned short* __restrict__ out, int K, int N) {
    const int total = K * N;
    for (int i = blockIdx.x * blockDim.x + threadIdx.x; i < total;
         i += gridDim.x * blockDim.x) {
        const int k = i / N;
        const int n = i - k * N;
        out[(size_t)n * K + k] = f2bf(in[i]);
    }
}

// ---------------------------------------------------------------------------
// bf16 WMMA GEMM: C[M,N] = A[M,K] * Bt[N,K]^T  (+ fused epilogue)
//   B is ALWAYS stored transposed: row n of Bt = column n of logical B,
//   with row stride ldb (in elements).
//   MODE 0: store bf16 (row-major, ldc)
//   MODE 1: store f32
//   MODE 2: store f32 = acc + bias[n] + resid[m*ldc+n]
//   MODE 3: store bf16 = relu(acc + bias[n])
//   MODE 4: Cf[m*ldc+n] += acc + bias[n]
//   MODE 5: store bf16 TRANSPOSED: Cbf[n*ldc + m]  (ldc = transposed stride)
//
// Block: 256 threads (8 wave32). Macro tile 128x64, BK=32.
// Each wave computes a 32x32 C block = 2x2 WMMA tiles (4 wmma / K-step).
// Double-buffered LDS staging via ASYNC global->LDS (3 async ops / thread
// per stage; s_wait_asynccnt 3 retires the previous stage).
// ---------------------------------------------------------------------------
#define ASTR 40   // padded LDS row stride (ushorts) = 80B, keeps 16B align
#define BSTR 40

template <int MODE>
__global__ __launch_bounds__(256) void gemm_bf16_wmma(
    const unsigned short* __restrict__ A,
    const unsigned short* __restrict__ Bt,
    float* __restrict__ Cf,
    unsigned short* __restrict__ Cbf,
    const float* __restrict__ bias,
    const float* __restrict__ resid,
    int M, int N, int Kd, int lda, int ldb, int ldc,
    size_t strideA, size_t strideB, size_t strideC) {
    const int bz = blockIdx.z;
    A  += (size_t)bz * strideA;
    Bt += (size_t)bz * strideB;
    if (Cf)  Cf  += (size_t)bz * strideC;
    if (Cbf) Cbf += (size_t)bz * strideC;

    const int tid  = threadIdx.x;
    const int lane = tid & 31;
    const int wave = tid >> 5;
    const int wm   = wave & 3;    // 0..3 -> 32-row band
    const int wn   = wave >> 2;   // 0..1 -> 32-col band
    const int m0   = blockIdx.y * 128;
    const int n0   = blockIdx.x * 64;

    __shared__ __align__(16) unsigned short As[2][128 * ASTR];
    __shared__ __align__(16) unsigned short Bs[2][64 * BSTR];

    FragF acc[2][2];
#pragma unroll
    for (int tm = 0; tm < 2; ++tm)
#pragma unroll
        for (int tn = 0; tn < 2; ++tn)
            acc[tm][tn].v = v8f{0.f, 0.f, 0.f, 0.f, 0.f, 0.f, 0.f, 0.f};

    // per-lane fragment addressing (CDNA5 wave32 WMMA layouts)
    const int arow0 = wm * 32 + (lane & 15);
    const int akoff = (lane >> 4) << 3;    // 0 / 8
    const int brow0 = wn * 32 + (lane & 15);
    const int bkoff = (lane >> 4) << 3;    // 0 / 8

    // A stage: 128 rows x 4 16B-chunks = 512 chunks -> 2 per thread
    // B stage:  64 rows x 4 16B-chunks = 256 chunks -> 1 per thread
    auto stage = [&](int kidx, int buf) {
        const int k0 = kidx * 32;
#pragma unroll
        for (int rep = 0; rep < 2; ++rep) {
            const int lin = rep * 256 + tid;
            const int r = lin >> 2;
            const int c = (lin & 3) << 3;
            copy16_g2lds(&A[(size_t)(m0 + r) * lda + k0 + c],
                         &As[buf][r * ASTR + c]);
        }
        {
            const int r = tid >> 2;
            const int c = (tid & 3) << 3;
            copy16_g2lds(&Bt[(size_t)(n0 + r) * ldb + k0 + c],
                         &Bs[buf][r * BSTR + c]);
        }
    };

    const int nk = Kd >> 5;
    stage(0, 0);
    for (int i = 0; i < nk; ++i) {
        const int cur = i & 1;
        if (i + 1 < nk) {
            stage(i + 1, cur ^ 1);
            wait_async_le3();      // previous stage's 3 ops retired
        } else {
            wait_async_le0();
        }
        __syncthreads();           // all waves' copies into buf[cur] visible

        FragBF a[2], b[2];
#pragma unroll
        for (int t = 0; t < 2; ++t) {
            const unsigned short* ap = &As[cur][(arow0 + 16 * t) * ASTR + akoff];
            *(uint4*)&a[t].u[0] = *(const uint4*)ap;
            *(uint4*)&a[t].u[4] = *(const uint4*)(ap + 16);
            const unsigned short* bp = &Bs[cur][(brow0 + 16 * t) * BSTR + bkoff];
            *(uint4*)&b[t].u[0] = *(const uint4*)bp;
            *(uint4*)&b[t].u[4] = *(const uint4*)(bp + 16);
        }
#pragma unroll
        for (int tm = 0; tm < 2; ++tm)
#pragma unroll
            for (int tn = 0; tn < 2; ++tn)
                acc[tm][tn].v = __builtin_amdgcn_wmma_f32_16x16x32_bf16(
                    false, a[tm].v, false, b[tn].v, (short)0, acc[tm][tn].v,
                    false, false);

        __syncthreads();           // buf[cur] free for reuse next iteration
    }

    // ---- epilogue ----
#pragma unroll
    for (int tm = 0; tm < 2; ++tm) {
#pragma unroll
        for (int tn = 0; tn < 2; ++tn) {
            const int mbase = m0 + wm * 32 + tm * 16 + ((lane >> 4) << 3);
            const int col   = n0 + wn * 32 + tn * 16 + (lane & 15);
            if (MODE == 5) {
                union { unsigned short s[8]; uint4 q; } pk;
#pragma unroll
                for (int v = 0; v < 8; ++v) pk.s[v] = f2bf(acc[tm][tn].f[v]);
                *(uint4*)&Cbf[(size_t)col * ldc + mbase] = pk.q;
            } else {
#pragma unroll
                for (int v = 0; v < 8; ++v) {
                    const size_t idx = (size_t)(mbase + v) * ldc + col;
                    float val = acc[tm][tn].f[v];
                    if (MODE == 0) {
                        Cbf[idx] = f2bf(val);
                    } else if (MODE == 1) {
                        Cf[idx] = val;
                    } else if (MODE == 2) {
                        Cf[idx] = val + bias[col] + resid[idx];
                    } else if (MODE == 3) {
                        val += bias[col];
                        Cbf[idx] = f2bf(val > 0.f ? val : 0.f);
                    } else if (MODE == 4) {
                        Cf[idx] += val + bias[col];
                    }
                }
            }
        }
    }
}

// ---------------------------------------------------------------------------
// Row softmax over T=2048 (one block / row, 256 threads, 8 elems/thread).
// In-place on f32 scores (applies scale), also emits bf16 copy.
// ---------------------------------------------------------------------------
__global__ __launch_bounds__(256) void softmax_row(
    float* __restrict__ S, unsigned short* __restrict__ Pbf,
    int T, float scale) {
    const size_t row = blockIdx.x;
    float* r = S + row * (size_t)T;
    unsigned short* p = Pbf + row * (size_t)T;
    const int tid = threadIdx.x;

    __shared__ float red[8];
    float x[8];

    float m = -3.4e38f;
#pragma unroll
    for (int i = 0; i < 8; ++i) {
        x[i] = r[tid + i * 256] * scale;
        m = fmaxf(m, x[i]);
    }
#pragma unroll
    for (int off = 16; off > 0; off >>= 1)
        m = fmaxf(m, __shfl_xor(m, off, 32));
    if ((tid & 31) == 0) red[tid >> 5] = m;
    __syncthreads();
    m = red[0];
#pragma unroll
    for (int w = 1; w < 8; ++w) m = fmaxf(m, red[w]);
    __syncthreads();

    float s = 0.f;
#pragma unroll
    for (int i = 0; i < 8; ++i) {
        x[i] = __expf(x[i] - m);
        s += x[i];
    }
#pragma unroll
    for (int off = 16; off > 0; off >>= 1)
        s += __shfl_xor(s, off, 32);
    if ((tid & 31) == 0) red[tid >> 5] = s;
    __syncthreads();
    s = red[0];
#pragma unroll
    for (int w = 1; w < 8; ++w) s += red[w];
    const float inv = 1.f / s;

#pragma unroll
    for (int i = 0; i < 8; ++i) {
        const float v = x[i] * inv;
        r[tid + i * 256] = v;
        p[tid + i * 256] = f2bf(v);
    }
}

// ---------------------------------------------------------------------------
// Host side
// ---------------------------------------------------------------------------
extern "C" void kernel_launch(void* const* d_in, const int* in_sizes, int n_in,
                              void* d_out, int out_size, void* d_ws,
                              size_t ws_size, hipStream_t stream) {
    (void)in_sizes; (void)n_in; (void)out_size; (void)ws_size;

    constexpr int B  = 4;
    constexpr int T  = 2048;
    constexpr int D  = 1024;
    constexpr int DI = 256;
    constexpr int MT = B * T;                      // 8192 rows
    constexpr size_t OUT_ELEMS = (size_t)MT * D;

    const float* X  = (const float*)d_in[0];
    const float* wq = (const float*)d_in[1];
    const float* wk = (const float*)d_in[2];
    const float* wv = (const float*)d_in[3];
    const float* wo = (const float*)d_in[4];
    const float* bo = (const float*)d_in[5];
    const float* w1 = (const float*)d_in[6];
    const float* b1 = (const float*)d_in[7];
    const float* w2 = (const float*)d_in[8];
    const float* b2 = (const float*)d_in[9];

    float* outF  = (float*)d_out;                  // (B,T,D) residual+ffn
    float* attnF = (float*)d_out + OUT_ELEMS;      // (B,T,T) attention

    // workspace layout (bf16 staging), all 16B aligned
    char* ws = (char*)d_ws;
    size_t off = 0;
    auto take = [&](size_t bytes) { char* p = ws + off; off += bytes; return p; };
    unsigned short* Xb   = (unsigned short*)take((size_t)MT * D * 2);
    unsigned short* WqT  = (unsigned short*)take((size_t)D * DI * 2);  // [DI][D]
    unsigned short* WkT  = (unsigned short*)take((size_t)D * DI * 2);
    unsigned short* WvT  = (unsigned short*)take((size_t)D * D * 2);   // [D][D]
    unsigned short* WoT  = (unsigned short*)take((size_t)D * D * 2);
    unsigned short* W1T  = (unsigned short*)take((size_t)D * DI * 2);  // [DI][D]
    unsigned short* W2T  = (unsigned short*)take((size_t)DI * D * 2);  // [D][DI]
    unsigned short* Qb   = (unsigned short*)take((size_t)MT * DI * 2); // [MT][DI]
    unsigned short* Kb   = (unsigned short*)take((size_t)MT * DI * 2); // [MT][DI]
    unsigned short* Vt   = (unsigned short*)take((size_t)MT * D * 2);  // [D][MT]
    unsigned short* Pb   = (unsigned short*)take((size_t)B * T * T * 2);
    unsigned short* Cb   = (unsigned short*)take((size_t)MT * D * 2);  // context
    unsigned short* Rb   = (unsigned short*)take((size_t)MT * D * 2);  // residual
    unsigned short* Hb   = (unsigned short*)take((size_t)MT * DI * 2); // hidden

    auto cvt = [&](const float* src, unsigned short* dst, int n) {
        int blocks = (n + 255) / 256;
        if (blocks > 8192) blocks = 8192;
        cvt_f32_bf16<<<blocks, 256, 0, stream>>>(src, dst, n);
    };
    auto cvtT = [&](const float* src, unsigned short* dst, int K, int N) {
        int n = K * N;
        int blocks = (n + 255) / 256;
        if (blocks > 8192) blocks = 8192;
        cvt_f32_bf16_T<<<blocks, 256, 0, stream>>>(src, dst, K, N);
    };

    // ---- stage inputs / weights (weights transposed to NxK) ----
    cvt(X, Xb, MT * D);
    cvtT(wq, WqT, D, DI);
    cvtT(wk, WkT, D, DI);
    cvtT(wv, WvT, D, D);
    cvtT(wo, WoT, D, D);
    cvtT(w1, W1T, D, DI);
    cvtT(w2, W2T, DI, D);

    // ---- Q = X@Wq ; K = X@Wk (row-major) ; V^T = (X@Wv)^T ----
    gemm_bf16_wmma<0><<<dim3(DI / 64, MT / 128, 1), 256, 0, stream>>>(
        Xb, WqT, nullptr, Qb, nullptr, nullptr, MT, DI, D, D, D, DI, 0, 0, 0);
    gemm_bf16_wmma<0><<<dim3(DI / 64, MT / 128, 1), 256, 0, stream>>>(
        Xb, WkT, nullptr, Kb, nullptr, nullptr, MT, DI, D, D, D, DI, 0, 0, 0);
    gemm_bf16_wmma<5><<<dim3(D / 64, MT / 128, 1), 256, 0, stream>>>(
        Xb, WvT, nullptr, Vt, nullptr, nullptr, MT, D, D, D, D, MT, 0, 0, 0);

    // ---- scores = Q @ K^T  (batched; K rows are already NxK) ----
    gemm_bf16_wmma<1><<<dim3(T / 64, T / 128, B), 256, 0, stream>>>(
        Qb, Kb, attnF, nullptr, nullptr, nullptr, T, T, DI, DI, DI, T,
        (size_t)T * DI, (size_t)T * DI, (size_t)T * T);

    // ---- attn = softmax(scores / 16), in place + bf16 copy ----
    softmax_row<<<B * T, 256, 0, stream>>>(attnF, Pb, T, 0.0625f);

    // ---- context = attn @ V  (batched; B operand = V^T rows) ----
    gemm_bf16_wmma<0><<<dim3(D / 64, T / 128, B), 256, 0, stream>>>(
        Pb, Vt, nullptr, Cb, nullptr, nullptr, T, D, T, T, MT, D,
        (size_t)T * T, (size_t)T, (size_t)T * D);

    // ---- residual = X + context @ Wo + bo  (f32, lives in d_out) ----
    gemm_bf16_wmma<2><<<dim3(D / 64, MT / 128, 1), 256, 0, stream>>>(
        Cb, WoT, outF, nullptr, bo, X, MT, D, D, D, D, D, 0, 0, 0);

    // ---- residual -> bf16 for FFN ----
    cvt(outF, Rb, MT * D);

    // ---- h = relu(residual @ W1 + b1) ----
    gemm_bf16_wmma<3><<<dim3(DI / 64, MT / 128, 1), 256, 0, stream>>>(
        Rb, W1T, nullptr, Hb, b1, nullptr, MT, DI, D, D, D, DI, 0, 0, 0);

    // ---- out = residual + h @ W2 + b2  (accumulate into d_out) ----
    gemm_bf16_wmma<4><<<dim3(D / 64, MT / 128, 1), 256, 0, stream>>>(
        Hb, W2T, outF, nullptr, b2, nullptr, MT, D, DI, DI, DI, D, 0, 0, 0);
}
